// MultiNL_55001351192940
// MI455X (gfx1250) — matrix operand order repaired and checked by reading, not compiled, not run on hardware
//
#include <hip/hip_runtime.h>
#include <hip/hip_bf16.h>

// MI455X / gfx1250: wave32, WMMA bf16 16x16x32 (f32 accum) for all GEMM stages,
// async global->LDS staging (ASYNCcnt) with double buffering in the hot GEMM.

typedef __attribute__((ext_vector_type(16))) __bf16 v16bf;
typedef __attribute__((ext_vector_type(8)))  __bf16 v8bf;
typedef __attribute__((ext_vector_type(8)))  float  v8f;

#define BN_EPS 1e-5f

__device__ __forceinline__ v8f wmma_bf16(v16bf a, v16bf b, v8f c) {
  // (neg_a, A, neg_b, B, c_mod, C, reuse_a, reuse_b)
  return __builtin_amdgcn_wmma_f32_16x16x32_bf16(false, a, false, b, (short)0, c,
                                                 false, false);
}

// A-matrix fragment (16x32 bf16): lane L<16 holds row M=L, K in {0..7,16..23};
// lane L>=16 holds row M=L-16, K in {8..15,24..31}. Two 16B LDS reads.
__device__ __forceinline__ v16bf load_a_frag(const __bf16* base, int row, int rs,
                                             int k0, int lane) {
  const int g8 = (lane < 16) ? 0 : 8;
  const __bf16* p = base + row * rs + k0 + g8;
  v8bf lo = *(const v8bf*)(p);
  v8bf hi = *(const v8bf*)(p + 16);
  return __builtin_shufflevector(lo, hi, 0, 1, 2, 3, 4, 5, 6, 7, 8, 9, 10, 11,
                                 12, 13, 14, 15);
}

// B-matrix fragment (32x16 bf16): lane = column N (mod 16), K same as A:
// lanes 0-15 -> K in {0..7,16..23}, lanes 16-31 -> K in {8..15,24..31}. Two
// 16B LDS reads with a [col][k] LDS layout.
__device__ __forceinline__ v16bf load_b_frag(const __bf16* base, int row, int rs,
                                             int k0, int lane) {
  const int g8 = (lane < 16) ? 0 : 8;
  const __bf16* p = base + row * rs + k0 + g8;
  v8bf lo = *(const v8bf*)(p);
  v8bf hi = *(const v8bf*)(p + 16);
  return __builtin_shufflevector(lo, hi, 0, 1, 2, 3, 4, 5, 6, 7, 8, 9, 10, 11,
                                 12, 13, 14, 15);
}

// Async DMA: 16 bytes global -> LDS per lane (tracked by ASYNCcnt).
// LDS operand VGPR = low 32 bits of the generic pointer (HW truncates to the
// wave's LDS offset, ISA 10.2 aperture rules).
__device__ __forceinline__ void async_copy_b128(const void* gsrc, void* ldst) {
  unsigned lds = (unsigned)(unsigned long long)ldst;
  unsigned long long ga = (unsigned long long)gsrc;
  asm volatile("global_load_async_to_lds_b128 %0, %1, off" ::"v"(lds), "v"(ga)
               : "memory");
}
__device__ __forceinline__ void wait_async0() {
  asm volatile("s_wait_asynccnt 0x0" ::: "memory");
}

// ---------------------------------------------------------------------------
// K0: collapse the w3-conv + wt dot:  vt[c] = sum_ic wt[ic]*w3[ic][c],
//     ct = wt . b3.  Then t[b,n] = vt . x[b,:,n] + ct.
// ---------------------------------------------------------------------------
__global__ void k0_fuse(const float* __restrict__ w3, const float* __restrict__ b3,
                        const float* __restrict__ w5, float* __restrict__ vt) {
  const int c = threadIdx.x;  // 128 threads
  float s = 0.f;
  for (int ic = 0; ic < 64; ++ic) s += w5[ic] * w3[ic * 128 + c];
  vt[c] = s;
  if (c == 0) {
    float ct = 0.f;
    for (int ic = 0; ic < 64; ++ic) ct += w5[ic] * b3[ic];
    vt[128] = ct;
  }
}

// ---------------------------------------------------------------------------
// K1: per (batch, pooled-row): stage x tile (2 rows x 64 cols = 128 px) into
// LDS transposed as bf16; WMMA-GEMM conv1 (w1) and conv4 (w4); 2x2 maxpool;
// write xg TRANSPOSED (bf16, [b][ic][m]) so K2 can async-DMA it; reduce
// p[b][m] = wp . pooled4; compute t via the collapsed vt vector.
// ---------------------------------------------------------------------------
__global__ __launch_bounds__(256) void k1_front(
    const float* __restrict__ x, const float* __restrict__ w1,
    const float* __restrict__ b1, const float* __restrict__ w4,
    const float* __restrict__ b4, const float* __restrict__ w5,
    const float* __restrict__ vt, float* __restrict__ t_out,
    float* __restrict__ p_out, __bf16* __restrict__ xg) {
  __shared__ __align__(32) __bf16 x_t[128 * 128];    // [px][c]  32KB
  __shared__ __align__(32) __bf16 w_lds[64 * 128];   // [ic][c]  16KB
  __shared__ __align__(32) __bf16 out_lds[64 * 128]; // [ic][px] 16KB

  const int tid = threadIdx.x;
  const int lane = tid & 31, wave = tid >> 5;
  const int prow = blockIdx.x;  // pooled row 0..31 -> input rows 2p, 2p+1
  const int b = blockIdx.y;
  const size_t xbase = ((size_t)b * 128) * 4096 + (size_t)prow * 128;

  for (int idx = tid; idx < 128 * 128; idx += 256) {
    int c = idx >> 7, px = idx & 127;
    x_t[px * 128 + c] = (__bf16)x[xbase + (size_t)c * 4096 + px];
  }
  for (int idx = tid; idx < 64 * 128; idx += 256) w_lds[idx] = (__bf16)w1[idx];
  __syncthreads();

  for (int pass = 0; pass < 2; ++pass) {
    // out[ic,px] = sum_c w[ic,c] x[c,px]; wave -> px-tile, 4 ic-tiles.
    v8f acc[4] = {};
    for (int k = 0; k < 4; ++k) {
      v16bf bf = load_b_frag(x_t, wave * 16 + (lane & 15), 128, 32 * k, lane);
#pragma unroll
      for (int j = 0; j < 4; ++j) {
        v16bf af = load_a_frag(w_lds, j * 16 + (lane & 15), 128, 32 * k, lane);
        acc[j] = wmma_bf16(af, bf, acc[j]);
      }
    }
    const float* bias = pass ? b4 : b1;
#pragma unroll
    for (int j = 0; j < 4; ++j) {
#pragma unroll
      for (int r = 0; r < 8; ++r) {
        int ic = j * 16 + r + ((lane < 16) ? 0 : 8);
        float v = acc[j][r] + bias[ic];
        out_lds[ic * 128 + wave * 16 + (lane & 15)] = (__bf16)v;
      }
    }
    __syncthreads();

    if (pass == 0) {
      // maxpool conv1 -> xg (transposed layout), stage w4 concurrently
      for (int idx = tid; idx < 64 * 32; idx += 256) {
        int ic = idx >> 5, j = idx & 31;
        float a0 = (float)out_lds[ic * 128 + 2 * j];
        float a1 = (float)out_lds[ic * 128 + 2 * j + 1];
        float a2 = (float)out_lds[ic * 128 + 64 + 2 * j];
        float a3 = (float)out_lds[ic * 128 + 64 + 2 * j + 1];
        float mx = fmaxf(fmaxf(a0, a1), fmaxf(a2, a3));
        xg[((size_t)b * 64 + ic) * 1024 + prow * 32 + j] = (__bf16)mx;
      }
      for (int idx = tid; idx < 64 * 128; idx += 256) w_lds[idx] = (__bf16)w4[idx];
      __syncthreads();
    }
  }

  // pass-1 pool + p reduction (overlay p_sh on freed w_lds), plus t from x_t.
  float* p_sh = reinterpret_cast<float*>(w_lds);
  if (tid < 32) p_sh[tid] = 0.f;
  const float ct = vt[128];
  __syncthreads();

  const float* wp = w5 + 64;
  for (int idx = tid; idx < 64 * 32; idx += 256) {
    int ic = idx >> 5, j = idx & 31;
    float a0 = (float)out_lds[ic * 128 + 2 * j];
    float a1 = (float)out_lds[ic * 128 + 2 * j + 1];
    float a2 = (float)out_lds[ic * 128 + 64 + 2 * j];
    float a3 = (float)out_lds[ic * 128 + 64 + 2 * j + 1];
    float mx = fmaxf(fmaxf(a0, a1), fmaxf(a2, a3));
    atomicAdd(&p_sh[j], wp[ic] * mx);
  }
  if (tid < 128) {
    float s = ct;
    for (int c = 0; c < 128; ++c) s += vt[c] * (float)x_t[tid * 128 + c];
    t_out[(size_t)b * 4096 + prow * 128 + tid] = s;
  }
  __syncthreads();
  if (tid < 32) p_out[b * 1024 + prow * 32 + tid] = p_sh[tid];
}

// ---------------------------------------------------------------------------
// K2: the heavy GEMM  y1[b,n,i] = (1/M) sum_m relu(t[n]+p[m]) * xg[m,i].
// feats A-fragments are synthesized in registers per k-step (never stored).
// xg (stored [b][ic][m]) is DMA'd per 128-m chunk via async-to-LDS with
// double buffering so the copy of chunk k+1 overlaps the WMMAs on chunk k.
// ---------------------------------------------------------------------------
__global__ __launch_bounds__(256) void k2_main(const float* __restrict__ t_in,
                                               const float* __restrict__ p_in,
                                               const __bf16* __restrict__ xg,
                                               __bf16* __restrict__ y1) {
  __shared__ __align__(32) __bf16 xgbuf[2][64 * 128];  // [i][m_local] 2x16KB
  __shared__ float p_sh[1024];                         // 4KB
  const int tid = threadIdx.x, lane = tid & 31, wave = tid >> 5;
  const int b = blockIdx.y;
  const int nbase = blockIdx.x * 128;  // 8 n-tiles, one per wave

  for (int idx = tid; idx < 1024; idx += 256) p_sh[idx] = p_in[b * 1024 + idx];
  const float tval = t_in[(size_t)b * 4096 + nbase + wave * 16 + (lane & 15)];
  const int g8 = (lane < 16) ? 0 : 8;

  const __bf16* xg_b = xg + (size_t)b * 64 * 1024;
  // Async-stage one 64x128 bf16 chunk (16KB): 1024 x 16B, 4 per thread.
  auto issue_chunk = [&](int kb, int buf) {
#pragma unroll
    for (int v = 0; v < 4; ++v) {
      int idx = v * 256 + tid;   // 0..1023
      int ic = idx >> 4;         // row (i)
      int m8 = idx & 15;         // 8-element group within the 128-m chunk
      async_copy_b128(xg_b + (size_t)ic * 1024 + kb * 128 + m8 * 8,
                      &xgbuf[buf][ic * 128 + m8 * 8]);
    }
  };

  issue_chunk(0, 0);
  v8f acc[4] = {};
  for (int kb = 0; kb < 8; ++kb) {  // m chunks of 128
    wait_async0();     // our chunk-kb DMAs complete
    __syncthreads();   // all waves' DMAs complete; prior compute drained
    if (kb + 1 < 8) issue_chunk(kb + 1, (kb + 1) & 1);  // overlap with WMMAs
    const __bf16* xgt = xgbuf[kb & 1];
    for (int ks = 0; ks < 4; ++ks) {
      const int kbase = kb * 128 + ks * 32;
      v16bf a;
#pragma unroll
      for (int e = 0; e < 16; ++e) {  // A lane layout K mapping
        int kk = ((e < 8) ? e : (e + 8)) + g8;
        float f = tval + p_sh[kbase + kk];
        a[e] = (__bf16)fmaxf(f, 0.f);
      }
#pragma unroll
      for (int it = 0; it < 4; ++it) {
        v16bf bf = load_b_frag(xgt, it * 16 + (lane & 15), 128, ks * 32, lane);
        acc[it] = wmma_bf16(a, bf, acc[it]);
      }
    }
  }
  const float sc = 1.0f / 1024.0f;
#pragma unroll
  for (int it = 0; it < 4; ++it) {
#pragma unroll
    for (int r = 0; r < 8; ++r) {
      int n = nbase + wave * 16 + r + ((lane < 16) ? 0 : 8);
      int i = it * 16 + (lane & 15);
      y1[((size_t)b * 4096 + n) * 64 + i] = (__bf16)(acc[it][r] * sc);
    }
  }
}

// ---------------------------------------------------------------------------
// K3: y = w2 @ y1 + b2, then fused BN + residual. K=64 -> 2 WMMA k-steps.
// y1 tile (16KB, contiguous) staged via async-to-LDS DMA.
// ---------------------------------------------------------------------------
__global__ __launch_bounds__(256) void k3_out(
    const __bf16* __restrict__ y1, const float* __restrict__ w2,
    const float* __restrict__ b2, const float* __restrict__ gamma,
    const float* __restrict__ beta, const float* __restrict__ mean,
    const float* __restrict__ var, const float* __restrict__ x,
    float* __restrict__ out) {
  __shared__ __align__(32) __bf16 w2_lds[128 * 64];  // [c][i] 16KB
  __shared__ __align__(32) __bf16 y1_sh[128 * 64];   // [n][i] 16KB
  const int tid = threadIdx.x, lane = tid & 31, wave = tid >> 5;
  const int b = blockIdx.y;
  const int nbase = blockIdx.x * 128;

  // async DMA the y1 tile (no conversion needed)
  const __bf16* y1_tile = y1 + ((size_t)b * 4096 + nbase) * 64;
#pragma unroll
  for (int v = 0; v < 4; ++v) {
    int idx = v * 256 + tid;  // 1024 x 16B
    async_copy_b128(y1_tile + idx * 8, &y1_sh[idx * 8]);
  }
  // w2 needs f32->bf16 conversion: regular path, overlaps the DMA above
  for (int idx = tid; idx < 128 * 64; idx += 256) w2_lds[idx] = (__bf16)w2[idx];
  wait_async0();
  __syncthreads();

  const int n = nbase + wave * 16 + (lane & 15);
  for (int ctile = 0; ctile < 8; ++ctile) {
    v8f acc = {};
#pragma unroll
    for (int ks = 0; ks < 2; ++ks) {
      v16bf af = load_a_frag(w2_lds, ctile * 16 + (lane & 15), 64, ks * 32, lane);
      v16bf bf = load_b_frag(y1_sh, wave * 16 + (lane & 15), 64, ks * 32, lane);
      acc = wmma_bf16(af, bf, acc);
    }
#pragma unroll
    for (int r = 0; r < 8; ++r) {
      int c = ctile * 16 + r + ((lane < 16) ? 0 : 8);
      float inv = rsqrtf(var[c] + BN_EPS);
      float v = acc[r] + b2[c];
      v = (v - mean[c]) * (gamma[c] * inv) + beta[c];
      size_t gi = ((size_t)b * 128 + c) * 4096 + n;
      out[gi] = v + x[gi];
    }
  }
}

// ---------------------------------------------------------------------------
extern "C" void kernel_launch(void* const* d_in, const int* in_sizes, int n_in,
                              void* d_out, int out_size, void* d_ws,
                              size_t ws_size, hipStream_t stream) {
  (void)in_sizes; (void)n_in; (void)out_size; (void)ws_size;
  const float* x     = (const float*)d_in[0];
  const float* w1    = (const float*)d_in[1];
  const float* b1    = (const float*)d_in[2];
  const float* w3    = (const float*)d_in[3];
  const float* b3    = (const float*)d_in[4];
  const float* w4    = (const float*)d_in[5];
  const float* b4    = (const float*)d_in[6];
  const float* w5    = (const float*)d_in[7];
  const float* w2    = (const float*)d_in[8];
  const float* b2    = (const float*)d_in[9];
  const float* gamma = (const float*)d_in[10];
  const float* beta  = (const float*)d_in[11];
  const float* mean  = (const float*)d_in[12];
  const float* var   = (const float*)d_in[13];
  float* out = (float*)d_out;

  // workspace layout (~5.2 MB)
  char* ws = (char*)d_ws;
  float*  vt    = (float*)(ws);                         // 129 f32 (pad to 1KB)
  float*  t_ws  = (float*)(ws + 1024);                  // 8*4096 f32
  float*  p_ws  = (float*)(ws + 1024 + 131072);         // 8*1024 f32
  __bf16* xg_ws = (__bf16*)(ws + 1024 + 131072 + 32768);            // 8*64*1024 bf16 (transposed)
  __bf16* y1_ws = (__bf16*)(ws + 1024 + 131072 + 32768 + 1048576);  // 8*4096*64 bf16

  k0_fuse<<<1, 128, 0, stream>>>(w3, b3, w5, vt);
  k1_front<<<dim3(32, 8), 256, 0, stream>>>(x, w1, b1, w4, b4, w5, vt, t_ws,
                                            p_ws, xg_ws);
  k2_main<<<dim3(32, 8), 256, 0, stream>>>(t_ws, p_ws, xg_ws, y1_ws);
  k3_out<<<dim3(32, 8), 256, 0, stream>>>(y1_ws, w2, b2, gamma, beta, mean, var,
                                          x, out);
}
